// STFT_83958020702337
// MI455X (gfx1250) — matrix operand order repaired and checked by reading, not compiled, not run on hardware
//
#include <hip/hip_runtime.h>
#include <hip/hip_bf16.h>
#include <math.h>

// ---------------- problem constants ----------------
#define NFFT   800
#define HOP    200
#define C2     802            // 2*(N_FFT/2+1) channels (re;im)
#define CPAD   832            // C2 padded to multiple of 32 (WMMA K, and fwd M)
#define NB     32             // batch
#define LSIG   160000
#define PADR   400            // N_FFT/2 reflect pad
#define XPLEN  (LSIG + 2*PADR)   // 160800
#define TFR    801            // frames
#define NCOL   (NB * TFR)     // 25632, multiple of 32

typedef __attribute__((ext_vector_type(16))) _Float16 v16h;
typedef __attribute__((ext_vector_type(8)))  _Float16 v8h;
typedef __attribute__((ext_vector_type(8)))  float    v8f;

union V16U { v16h v; v8h h[2]; };

// ---------------- workspace layout (bytes) ----------------
// xp16   : NB*XPLEN f16
// fb16   : CPAD*NFFT f16 (A of GEMM1, rows>=802 zero)
// invT16 : NFFT*CPAD f16 (A of GEMM2, pre-transposed [k][c])
// wssq   : XPLEN f32
// recomb : NCOL*CPAD f16 ([n][c], B of GEMM2; fully written by GEMM1)
// acc    : NB*XPLEN f32  (overlap-add accumulator, zeroed per call)
#define OFF_XP   ((size_t)0)
#define OFF_FB   ((size_t)NB*XPLEN*2)                     // 10,291,200
#define OFF_INV  (OFF_FB  + (size_t)CPAD*NFFT*2)
#define OFF_WS   (OFF_INV + (size_t)NFFT*CPAD*2)
#define OFF_RC   (((OFF_WS + (size_t)XPLEN*4 + 255)/256)*256)
#define OFF_ACC  (OFF_RC  + (size_t)NCOL*CPAD*2)

__device__ __forceinline__ float fb_raw(int c, int k) {
    // raw stacked DFT: c<=400 -> cos(2*pi*c*k/N); c>=401 -> -sin(2*pi*(c-401)*k/N)
    int m = (c <= 400) ? c : (c - 401);
    int r = (m * k) % NFFT;                       // exact integer phase reduction
    float th = 6.28318530717958647692f * (float)r / (float)NFFT;
    return (c <= 400) ? __cosf(th) : -__sinf(th);
}
__device__ __forceinline__ float hann_w(int k) {
    float th = 6.28318530717958647692f * (float)k / (float)NFFT;
    return 0.5f - 0.5f * __cosf(th);
}

// ---------------- setup kernels ----------------
__global__ void k_zero_u4(uint4* __restrict__ p, long n4) {
    long i = (long)blockIdx.x * blockDim.x + threadIdx.x;
    if (i < n4) p[i] = make_uint4(0u, 0u, 0u, 0u);
}

__global__ void k_build_xp(const float* __restrict__ x, _Float16* __restrict__ xp16) {
    long i = (long)blockIdx.x * blockDim.x + threadIdx.x;
    if (i >= (long)NB * XPLEN) return;
    int b = (int)(i / XPLEN), p = (int)(i % XPLEN);
    int j = p - PADR;
    if (j < 0) j = -j;                            // numpy 'reflect'
    if (j >= LSIG) j = 2 * LSIG - 2 - j;
    xp16[i] = (_Float16)x[(size_t)b * LSIG + j];
}

__global__ void k_build_fb(_Float16* __restrict__ fb) {
    long i = (long)blockIdx.x * blockDim.x + threadIdx.x;
    if (i >= (long)CPAD * NFFT) return;
    int c = (int)(i / NFFT), k = (int)(i % NFFT);
    float v = (c >= C2) ? 0.0f : fb_raw(c, k) * hann_w(k);
    fb[i] = (_Float16)v;
}

__global__ void k_build_inv(_Float16* __restrict__ invT) {
    long i = (long)blockIdx.x * blockDim.x + threadIdx.x;
    if (i >= (long)NFFT * CPAD) return;
    int k = (int)(i / CPAD), c = (int)(i % CPAD);
    float v = 0.0f;
    if (c < C2) {
        // pinv(4*fb).T * hann : DFT rows orthogonal -> (1/4) * fb_raw/||row||^2 * w
        int m = (c <= 400) ? c : (c - 401);
        float d = (m == 0 || m == 400) ? 800.0f : 400.0f;   // imag m=0/400 rows are 0 anyway
        v = 0.25f * fb_raw(c, k) * hann_w(k) / d;
    }
    invT[i] = (_Float16)v;
}

__global__ void k_build_wssq(float* __restrict__ wssq) {
    int n = blockIdx.x * blockDim.x + threadIdx.x;
    if (n >= XPLEN) return;
    int tmin = (n >= NFFT) ? ((n - NFFT) / HOP + 1) : 0;
    int tmax = n / HOP; if (tmax > TFR - 1) tmax = TFR - 1;
    float s = 0.0f;
    for (int t = tmin; t <= tmax; ++t) {
        float w = hann_w(n - t * HOP);
        s += w * w;
    }
    wssq[n] = s;
}

// ---------------- GEMM1 (2x2 blocked): ft[c,n] = sum_k FB[c,k]*frame[n][k] ----------------
// (mag/phase recombination is the identity: mag*cos(atan2(im,re)) == re)
// Each wave: 32x32 output tile = 4 WMMAs per K-step, A/B subtiles reused 2x each.
__global__ void __launch_bounds__(256)
k_gemm_fwd(const _Float16* __restrict__ xp16, const _Float16* __restrict__ fb,
           _Float16* __restrict__ recomb) {
    const int CS = CPAD / 32, NS = NCOL / 32;              // 26 x 801 supertiles
    int lane = threadIdx.x & 31;
    int tile = blockIdx.x * 8 + (threadIdx.x >> 5);
    if (tile >= CS * NS) return;                           // wave-uniform: EXEC all-ones
    int c0 = (tile % CS) * 32, n0 = (tile / CS) * 32;
    int lrow = lane & 15, hi = lane >> 4;

    const _Float16* aRow0 = fb + (size_t)(c0 + lrow) * NFFT;       // A rows, subtile 0
    const _Float16* aRow1 = aRow0 + (size_t)16 * NFFT;             // A rows, subtile 1
    int na = n0 + lrow, nb = n0 + 16 + lrow;
    int ba = na / TFR, ta = na % TFR;
    int bb = nb / TFR, tb = nb % TFR;
    const _Float16* bCol0 = xp16 + (size_t)ba * XPLEN + (size_t)ta * HOP;
    const _Float16* bCol1 = xp16 + (size_t)bb * XPLEN + (size_t)tb * HOP;

    v8f a00 = {}, a01 = {}, a10 = {}, a11 = {};
    for (int kt = 0; kt < NFFT / 32; ++kt) {
        int kb = kt * 32;
        V16U A0, A1, B0, B1;
        A0.h[0] = *(const v8h*)(aRow0 + kb + hi * 8);
        A0.h[1] = *(const v8h*)(aRow0 + kb + hi * 8 + 16);
        A1.h[0] = *(const v8h*)(aRow1 + kb + hi * 8);
        A1.h[1] = *(const v8h*)(aRow1 + kb + hi * 8 + 16);
        B0.h[0] = *(const v8h*)(bCol0 + kb + hi * 8);
        B0.h[1] = *(const v8h*)(bCol0 + kb + hi * 8 + 16);
        B1.h[0] = *(const v8h*)(bCol1 + kb + hi * 8);
        B1.h[1] = *(const v8h*)(bCol1 + kb + hi * 8 + 16);
        a00 = __builtin_amdgcn_wmma_f32_16x16x32_f16(false, A0.v, false, B0.v, (short)0, a00, false, false);
        a01 = __builtin_amdgcn_wmma_f32_16x16x32_f16(false, A0.v, false, B1.v, (short)0, a01, false, false);
        a10 = __builtin_amdgcn_wmma_f32_16x16x32_f16(false, A1.v, false, B0.v, (short)0, a10, false, false);
        a11 = __builtin_amdgcn_wmma_f32_16x16x32_f16(false, A1.v, false, B1.v, (short)0, a11, false, false);
    }
    // D element (c0 + 16*cs + j + 8*hi, n0 + 16*ns + lrow) -> recomb[n][c] (transposed store)
    v8h o00, o01, o10, o11;
    #pragma unroll
    for (int j = 0; j < 8; ++j) {
        o00[j] = (_Float16)a00[j]; o01[j] = (_Float16)a01[j];
        o10[j] = (_Float16)a10[j]; o11[j] = (_Float16)a11[j];
    }
    _Float16* r0 = recomb + (size_t)na * CPAD + c0 + hi * 8;
    _Float16* r1 = recomb + (size_t)nb * CPAD + c0 + hi * 8;
    *(v8h*)(r0)      = o00;
    *(v8h*)(r1)      = o01;
    *(v8h*)(r0 + 16) = o10;
    *(v8h*)(r1 + 16) = o11;
}

// ------- GEMM2 (2x2 blocked) + fused overlap-add: y[m,k] = sum_c recomb[m,c]*INVt[k,c] -------
__global__ void __launch_bounds__(256)
k_gemm_inv(const _Float16* __restrict__ invT, const _Float16* __restrict__ recomb,
           float* __restrict__ acc) {
    const int KS = NFFT / 32, MS = NCOL / 32;              // 25 x 801 supertiles
    int lane = threadIdx.x & 31;
    int tile = blockIdx.x * 8 + (threadIdx.x >> 5);
    if (tile >= KS * MS) return;
    int k0 = (tile % KS) * 32, m0 = (tile / KS) * 32;
    int lrow = lane & 15, hi = lane >> 4;

    const _Float16* aRow0 = invT + (size_t)(k0 + lrow) * CPAD;     // A: [k][c]
    const _Float16* aRow1 = aRow0 + (size_t)16 * CPAD;
    const _Float16* bCol0 = recomb + (size_t)(m0 + lrow) * CPAD;   // B: [m][c]
    const _Float16* bCol1 = bCol0 + (size_t)16 * CPAD;

    v8f a00 = {}, a01 = {}, a10 = {}, a11 = {};
    for (int ct = 0; ct < CPAD / 32; ++ct) {
        int cb = ct * 32;
        V16U A0, A1, B0, B1;
        A0.h[0] = *(const v8h*)(aRow0 + cb + hi * 8);
        A0.h[1] = *(const v8h*)(aRow0 + cb + hi * 8 + 16);
        A1.h[0] = *(const v8h*)(aRow1 + cb + hi * 8);
        A1.h[1] = *(const v8h*)(aRow1 + cb + hi * 8 + 16);
        B0.h[0] = *(const v8h*)(bCol0 + cb + hi * 8);
        B0.h[1] = *(const v8h*)(bCol0 + cb + hi * 8 + 16);
        B1.h[0] = *(const v8h*)(bCol1 + cb + hi * 8);
        B1.h[1] = *(const v8h*)(bCol1 + cb + hi * 8 + 16);
        a00 = __builtin_amdgcn_wmma_f32_16x16x32_f16(false, A0.v, false, B0.v, (short)0, a00, false, false);
        a01 = __builtin_amdgcn_wmma_f32_16x16x32_f16(false, A0.v, false, B1.v, (short)0, a01, false, false);
        a10 = __builtin_amdgcn_wmma_f32_16x16x32_f16(false, A1.v, false, B0.v, (short)0, a10, false, false);
        a11 = __builtin_amdgcn_wmma_f32_16x16x32_f16(false, A1.v, false, B1.v, (short)0, a11, false, false);
    }
    // overlap-add: D element (k0+16*ks+j+8*hi, m0+16*ms+lrow) adds into acc[b][t*HOP + k]
    int ma = m0 + lrow, mb = m0 + 16 + lrow;
    int bba = ma / TFR, tta = ma % TFR;
    int bbb = mb / TFR, ttb = mb % TFR;
    float* d0 = acc + (size_t)bba * XPLEN + (size_t)tta * HOP + k0 + hi * 8;
    float* d1 = acc + (size_t)bbb * XPLEN + (size_t)ttb * HOP + k0 + hi * 8;
    #pragma unroll
    for (int j = 0; j < 8; ++j) {
        unsafeAtomicAdd(d0 + j,      a00[j]);
        unsafeAtomicAdd(d1 + j,      a01[j]);
        unsafeAtomicAdd(d0 + 16 + j, a10[j]);
        unsafeAtomicAdd(d1 + 16 + j, a11[j]);
    }
}

// ---------------- finalize: window-sumsquare normalize, scale, slice pad ----------------
__global__ void k_finalize(const float* __restrict__ acc, const float* __restrict__ wssq,
                           float* __restrict__ out) {
    long i = (long)blockIdx.x * blockDim.x + threadIdx.x;
    if (i >= (long)NB * LSIG) return;
    int b = (int)(i / LSIG), n0 = (int)(i % LSIG);
    int n = n0 + PADR;
    float v = acc[(size_t)b * XPLEN + n];
    float w = wssq[n];
    out[i] = (w > 1e-5f ? v / w : v) * 4.0f;               // * (N_FFT/HOP)
}

// ---------------- launcher ----------------
extern "C" void kernel_launch(void* const* d_in, const int* in_sizes, int n_in,
                              void* d_out, int out_size, void* d_ws, size_t ws_size,
                              hipStream_t stream) {
    (void)in_sizes; (void)n_in; (void)out_size; (void)ws_size;
    const float* x  = (const float*)d_in[0];
    float* out      = (float*)d_out;
    char* ws        = (char*)d_ws;

    _Float16* xp16   = (_Float16*)(ws + OFF_XP);
    _Float16* fb16   = (_Float16*)(ws + OFF_FB);
    _Float16* invT16 = (_Float16*)(ws + OFF_INV);
    float*    wssq   = (float*)   (ws + OFF_WS);
    _Float16* recomb = (_Float16*)(ws + OFF_RC);
    float*    acc    = (float*)   (ws + OFF_ACC);

    const int TPB = 256;
    // zero only the OLA accumulator (recomb is fully overwritten by GEMM1 each call)
    long n4 = ((long)NB * XPLEN * 4) / 16;
    hipLaunchKernelGGL(k_zero_u4, dim3((unsigned)((n4 + TPB - 1) / TPB)), dim3(TPB), 0, stream,
                       (uint4*)(ws + OFF_ACC), n4);

    long nxp = (long)NB * XPLEN;
    hipLaunchKernelGGL(k_build_xp, dim3((unsigned)((nxp + TPB - 1) / TPB)), dim3(TPB), 0, stream, x, xp16);
    long nfb = (long)CPAD * NFFT;
    hipLaunchKernelGGL(k_build_fb, dim3((unsigned)((nfb + TPB - 1) / TPB)), dim3(TPB), 0, stream, fb16);
    long ninv = (long)NFFT * CPAD;
    hipLaunchKernelGGL(k_build_inv, dim3((unsigned)((ninv + TPB - 1) / TPB)), dim3(TPB), 0, stream, invT16);
    hipLaunchKernelGGL(k_build_wssq, dim3((XPLEN + TPB - 1) / TPB), dim3(TPB), 0, stream, wssq);

    int tiles1 = (CPAD / 32) * (NCOL / 32);                // 26*801
    hipLaunchKernelGGL(k_gemm_fwd, dim3((tiles1 + 7) / 8), dim3(TPB), 0, stream, xp16, fb16, recomb);

    int tiles2 = (NFFT / 32) * (NCOL / 32);                // 25*801
    hipLaunchKernelGGL(k_gemm_inv, dim3((tiles2 + 7) / 8), dim3(TPB), 0, stream, invT16, recomb, acc);

    long nout = (long)NB * LSIG;
    hipLaunchKernelGGL(k_finalize, dim3((unsigned)((nout + TPB - 1) / TPB)), dim3(TPB), 0, stream,
                       acc, wssq, out);
}
